// RNN_67542655696944
// MI455X (gfx1250) — matrix operand order repaired and checked by reading, not compiled, not run on hardware
//
#include <hip/hip_runtime.h>

#define FEATURE 512
#define HIDDEN  512
#define BSZ     64
#define T_STEPS 512
#define KDIM    (HIDDEN + FEATURE)   // 1024

typedef __attribute__((ext_vector_type(16))) __bf16 v16bf;
typedef __attribute__((ext_vector_type(8)))  float  v8f;

__device__ __forceinline__ unsigned short f2bf(float f) {
    // round-to-nearest-even fp32 -> bf16
    unsigned int u = __float_as_uint(f);
    unsigned int r = (u + 0x7FFFu + ((u >> 16) & 1u)) >> 16;
    return (unsigned short)r;
}

// ---- one-time prep: Wt[n][k] = bf16(W[k][n]); W: [KDIM,HIDDEN] fp32 ----
__global__ void prep_weight(const float* __restrict__ W,
                            unsigned short* __restrict__ Wt) {
    int idx = blockIdx.x * blockDim.x + threadIdx.x;   // over HIDDEN*KDIM
    int n = idx >> 10;          // / KDIM
    int k = idx & (KDIM - 1);   // % KDIM
    Wt[idx] = f2bf(W[k * HIDDEN + n]);
}

// ---- one-time prep: xbf = bf16(x), contiguous, 4 elems/thread ----
__global__ void prep_x(const float* __restrict__ x,
                       unsigned short* __restrict__ xbf) {
    long idx = (long)blockIdx.x * blockDim.x + threadIdx.x;
    float4 v = ((const float4*)x)[idx];
    unsigned p0 = (unsigned)f2bf(v.x) | ((unsigned)f2bf(v.y) << 16);
    unsigned p1 = (unsigned)f2bf(v.z) | ((unsigned)f2bf(v.w) << 16);
    ((uint2*)xbf)[idx] = make_uint2(p0, p1);
}

// 64B contiguous global(bf16) -> LDS copy, ASYNCcnt-tracked.
// ISA: INST_OFFSET is added to BOTH the LDS and the global address.
__device__ __forceinline__ void async_copy64(const unsigned short* g,
                                             unsigned lds_off) {
    asm volatile(
        "global_load_async_to_lds_b128 %0, %1, off\n\t"
        "global_load_async_to_lds_b128 %0, %1, off offset:16\n\t"
        "global_load_async_to_lds_b128 %0, %1, off offset:32\n\t"
        "global_load_async_to_lds_b128 %0, %1, off offset:48"
        :: "v"(lds_off), "v"(g) : "memory");
}

__device__ __forceinline__ void wait_async0() {
    asm volatile("s_wait_asynccnt 0x0" ::: "memory");
}

template <bool ASYNC>
__global__ void __launch_bounds__(256, 1)
rnn_fused(const float* __restrict__ x, const unsigned short* __restrict__ xbf,
          const float* __restrict__ h0, const unsigned short* __restrict__ Wt,
          const float* __restrict__ bias,
          float* __restrict__ out, float* __restrict__ hlast) {
    // Recurrent half of A (rewritten in place each step) + double-buffered x tile
    __shared__ __align__(16) unsigned short Ah[16 * HIDDEN];        // 16 KB
    __shared__ __align__(16) unsigned short Xb[2][16 * FEATURE];    // 32 KB

    const int tid  = threadIdx.x;
    const int lane = tid & 31;          // wave32
    const int wid  = tid >> 5;          // 0..7 waves
    const int b0   = blockIdx.x * 16;   // batch base for this WG
    const int m16  = lane & 15;
    const int half = lane >> 4;         // 0 or 1

    // per-thread staging coordinates: 8192 elems / 256 thr = 32 each
    const int e  = tid * 32;
    const int sm = e >> 9;              // row 0..15
    const int sf = e & 511;             // col, multiple of 32

    // ---- stage h0 -> Ah (fp32 -> bf16) ----
    {
        const float* src = h0 + (b0 + sm) * HIDDEN + sf;
        unsigned short* dst = &Ah[sm * HIDDEN + sf];
        #pragma unroll
        for (int i = 0; i < 8; ++i) {
            float4 v = ((const float4*)src)[i];
            unsigned p0 = (unsigned)f2bf(v.x) | ((unsigned)f2bf(v.y) << 16);
            unsigned p1 = (unsigned)f2bf(v.z) | ((unsigned)f2bf(v.w) << 16);
            ((uint2*)dst)[i] = make_uint2(p0, p1);
        }
    }

    if (ASYNC) {
        // prime the pipeline: async-load x_0 into Xb[0]
        const unsigned short* g =
            xbf + ((long)(b0 + sm) * T_STEPS + 0) * FEATURE + sf;
        async_copy64(g, (unsigned)(size_t)&Xb[0][sm * FEATURE + sf]);
    }

    for (int t = 0; t < T_STEPS; ++t) {
        const int xb = ASYNC ? (t & 1) : 0;

        if (ASYNC) {
            wait_async0();              // this wave's x_t copy landed in LDS
        } else {
            // fallback: convert-stage x_t into Xb[0] with VALU
            const float* src =
                x + ((long)(b0 + sm) * T_STEPS + t) * FEATURE + sf;
            unsigned short* dst = &Xb[0][sm * FEATURE + sf];
            #pragma unroll
            for (int i = 0; i < 8; ++i) {
                float4 v = ((const float4*)src)[i];
                unsigned p0 = (unsigned)f2bf(v.x) | ((unsigned)f2bf(v.y) << 16);
                unsigned p1 = (unsigned)f2bf(v.z) | ((unsigned)f2bf(v.w) << 16);
                ((uint2*)dst)[i] = make_uint2(p0, p1);
            }
            if (t + 1 < T_STEPS) __builtin_prefetch(src + FEATURE, 0, 0);
        }
        __syncthreads();   // A fully built (all waves) before any GEMM read

        if (ASYNC && (t + 1 < T_STEPS)) {
            // overlap next step's x staging with this step's GEMM
            const unsigned short* g =
                xbf + ((long)(b0 + sm) * T_STEPS + (t + 1)) * FEATURE + sf;
            async_copy64(g,
                (unsigned)(size_t)&Xb[(t + 1) & 1][sm * FEATURE + sf]);
        }

        // ---- GEMM: [16,1024] x [1024, 64-per-wave], bf16 WMMA, f32 acc ----
        v8f acc[4] = {};
        const unsigned short* Ab = &Ah[m16 * HIDDEN + half * 16];
        const unsigned short* Xp = &Xb[xb][m16 * FEATURE + half * 16];

        // phase 0: K = 0..511 (recurrent h)
        #pragma unroll 4
        for (int ks = 0; ks < 16; ++ks) {
            union { uint4 u[2]; v16bf v; } af;
            af.u[0] = *((const uint4*)(Ab + ks * 32));
            af.u[1] = *((const uint4*)(Ab + ks * 32 + 8));
            #pragma unroll
            for (int j = 0; j < 4; ++j) {
                int col = (wid * 4 + j) * 16 + m16;
                const unsigned short* bp =
                    Wt + (long)col * KDIM + ks * 32 + half * 16;
                union { uint4 u[2]; v16bf v; } bfrag;
                bfrag.u[0] = ((const uint4*)bp)[0];
                bfrag.u[1] = ((const uint4*)bp)[1];
                acc[j] = __builtin_amdgcn_wmma_f32_16x16x32_bf16(
                    false, af.v, false, bfrag.v, (short)0, acc[j], false, false);
            }
        }
        // phase 1: K = 512..1023 (x_t)
        #pragma unroll 4
        for (int ks = 0; ks < 16; ++ks) {
            union { uint4 u[2]; v16bf v; } af;
            af.u[0] = *((const uint4*)(Xp + ks * 32));
            af.u[1] = *((const uint4*)(Xp + ks * 32 + 8));
            #pragma unroll
            for (int j = 0; j < 4; ++j) {
                int col = (wid * 4 + j) * 16 + m16;
                const unsigned short* bp =
                    Wt + (long)col * KDIM + HIDDEN + ks * 32 + half * 16;
                union { uint4 u[2]; v16bf v; } bfrag;
                bfrag.u[0] = ((const uint4*)bp)[0];
                bfrag.u[1] = ((const uint4*)bp)[1];
                acc[j] = __builtin_amdgcn_wmma_f32_16x16x32_bf16(
                    false, af.v, false, bfrag.v, (short)0, acc[j], false, false);
            }
        }
        __syncthreads();   // all waves done reading Ah/Xb before overwrite

        // ---- bias + ReLU, store out (fp32), feed h_new back into Ah ----
        #pragma unroll
        for (int j = 0; j < 4; ++j) {
            int col = (wid * 4 + j) * 16 + m16;   // N index for this lane
            float bv = bias[col];
            #pragma unroll
            for (int v = 0; v < 8; ++v) {
                int mrow = v + 8 * half;          // M index (C/D layout)
                float r = fmaxf(acc[j][v] + bv, 0.0f);
                out[((long)(b0 + mrow) * T_STEPS + t) * HIDDEN + col] = r;
                Ah[mrow * HIDDEN + col] = f2bf(r);
                if (t == T_STEPS - 1)
                    hlast[(b0 + mrow) * HIDDEN + col] = r;
            }
        }
        // next iteration's barrier orders Ah/Xb rewrites vs. GEMM reads
    }
}

extern "C" void kernel_launch(void* const* d_in, const int* in_sizes, int n_in,
                              void* d_out, int out_size, void* d_ws, size_t ws_size,
                              hipStream_t stream) {
    (void)in_sizes; (void)n_in; (void)out_size;
    const float* x    = (const float*)d_in[0];   // [64, 512, 512]
    const float* h0   = (const float*)d_in[1];   // [64, 1, 512]
    const float* W    = (const float*)d_in[2];   // [1024, 512]
    const float* bias = (const float*)d_in[3];   // [512]

    float* out   = (float*)d_out;                              // [64,512,512]
    float* hlast = out + (size_t)BSZ * T_STEPS * HIDDEN;       // [64,1,512]

    unsigned short* Wt  = (unsigned short*)d_ws;               // 1 MB
    unsigned short* xbf = Wt + (size_t)HIDDEN * KDIM;          // 32 MB

    const size_t need = ((size_t)HIDDEN * KDIM +
                         (size_t)BSZ * T_STEPS * FEATURE) * sizeof(unsigned short);

    prep_weight<<<(HIDDEN * KDIM) / 256, 256, 0, stream>>>(W, Wt);

    if (ws_size >= need) {
        const long nx4 = (long)BSZ * T_STEPS * FEATURE / 4;
        prep_x<<<(unsigned)(nx4 / 256), 256, 0, stream>>>(x, xbf);
        rnn_fused<true><<<BSZ / 16, 256, 0, stream>>>(x, xbf, h0, Wt, bias, out, hlast);
    } else {
        rnn_fused<false><<<BSZ / 16, 256, 0, stream>>>(x, nullptr, h0, Wt, bias, out, hlast);
    }
}
